// encoder_model_22127671509464
// MI455X (gfx1250) — matrix operand order repaired and checked by reading, not compiled, not run on hardware
//
#include <hip/hip_runtime.h>
#include <hip/hip_bf16.h>

typedef __attribute__((ext_vector_type(16))) _Float16 v16h;
typedef __attribute__((ext_vector_type(8)))  _Float16 v8h;
typedef __attribute__((ext_vector_type(8)))  float    v8f;

#define BB 16384
#define TT 64
#define HH 14
#define WPB 4   // waves per block (128 threads)

static __device__ __forceinline__ float fast_tanh(float x) {
#if __has_builtin(__builtin_amdgcn_tanhf)
  return __builtin_amdgcn_tanhf(x);        // native v_tanh_f32 on gfx1250
#else
  return 1.0f - 2.0f / (__expf(2.0f * x) + 1.0f);
#endif
}
// sigmoid(x) = 0.5 + 0.5*tanh(x/2): one TRANS op instead of exp + precise-div chain
static __device__ __forceinline__ float fast_sigmoid(float x) {
  return __builtin_fmaf(0.5f, fast_tanh(0.5f * x), 0.5f);
}

// One LSTM pass over T=64 timesteps. Each wave handles a 16-row batch tile.
// IN  : input channels (13 for layer1, 28 for layers 2/3)
// KCH : number of 32-wide K chunks for the concat [x_t | h | 1] (1 or 2)
template<int IN, int KCH, bool SRC_F32, bool DST_F32>
__global__ __launch_bounds__(WPB * 32)
void lstm_pass(const void* __restrict__ srcv, int srcStride,
               const float* __restrict__ wih, const float* __restrict__ whh,
               const float* __restrict__ bih, const float* __restrict__ bhh,
               void* __restrict__ dstv, int dstStride, int dstOff,
               int reverse, int nTiles)
{
  constexpr int K     = IN + HH;        // real contraction depth
  constexpr int BIASK = KCH * 32 - 1;   // virtual bias channel (always 1.0)
  const int lane  = threadIdx.x & 31;
  const int wv    = threadIdx.x >> 5;
  const int tile  = blockIdx.x * WPB + wv;
  if (tile >= nTiles) return;           // wave-uniform: EXEC stays all-ones
  const int row0  = tile * 16;
  const int col   = lane & 15;          // N column (C/D layout) / M row (A layout)
  const int khalf = lane >> 4;          // lane half selects K sub-range / M+8

  // per-wave concat staging buffer: [row 0..15][K 0..63], f16
  __shared__ __align__(16) _Float16 smem[WPB][16][64];
  _Float16 (*a)[64] = smem[wv];

  // zero once (covers K padding columns), then plant the bias "1" channel.
  for (int i = lane; i < 16 * 64 / 2; i += 32)
    ((unsigned int*)&smem[wv][0][0])[i] = 0u;
  if (lane < 16) a[lane][BIASK] = (_Float16)1.0f;

  // ---- Build resident B fragments: one 16-col tile per gate, per K chunk ----
  // B 32x16 f16, this code's contiguous k order (same as A): lane n<16 holds col n, K=0..15 (pairs per VGPR);
  // lanes 16-31 hold K=16..31. Row BIASK carries the combined bias.
  v16h Bf[KCH][4];
  #pragma unroll
  for (int kc = 0; kc < KCH; ++kc) {
    #pragma unroll
    for (int g = 0; g < 4; ++g) {
      v16h bb;
      #pragma unroll
      for (int kk = 0; kk < 16; ++kk) {
        const int k = kc * 32 + khalf * 16 + kk;
        float w = 0.0f;
        if (col < HH) {
          if (k < IN)            w = wih[(g * HH + col) * IN + k];
          else if (k < K)        w = whh[(g * HH + col) * HH + (k - IN)];
          else if (k == BIASK)   w = bih[g * HH + col] + bhh[g * HH + col];
        }
        bb[kk] = (_Float16)w;
      }
      Bf[kc][g] = bb;
    }
  }

  const float*    srcF = (const float*)srcv;
  const _Float16* srcH = (const _Float16*)srcv;
  float*          dstF = (float*)dstv;
  _Float16*       dstH = (_Float16*)dstv;

  v8f c_st, h_st, zeroC;
  #pragma unroll
  for (int j = 0; j < 8; ++j) { c_st[j] = 0.0f; h_st[j] = 0.0f; zeroC[j] = 0.0f; }

  for (int s = 0; s < TT; ++s) {
    const int t = reverse ? (TT - 1 - s) : s;

    // prefetch next timestep's input rows (global_prefetch_b8 path)
    if (s + 1 < TT) {
      const int tn = reverse ? (TT - 2 - s) : (s + 1);
      const size_t pg = ((size_t)row0 * TT + tn) * (size_t)srcStride;
      __builtin_prefetch(SRC_F32 ? (const void*)&srcF[pg] : (const void*)&srcH[pg], 0, 1);
    }

    // h (C-layout f32 regs) -> LDS concat columns [IN .. IN+15]
    // lane holds (M = 8*khalf + j, N = col); cols 14/15 hit zero B rows -> harmless
    #pragma unroll
    for (int j = 0; j < 8; ++j)
      a[khalf * 8 + j][IN + col] = (_Float16)h_st[j];

    // x_t tile -> LDS concat columns [0 .. IN)
    for (int i = lane; i < 16 * IN; i += 32) {
      const int r = i / IN, ch = i % IN;
      const size_t gidx = ((size_t)(row0 + r) * TT + t) * (size_t)srcStride + ch;
      const float v = SRC_F32 ? srcF[gidx] : (float)srcH[gidx];
      a[r][ch] = (_Float16)v;
    }
    asm volatile("" ::: "memory");   // keep ds stores before ds loads

    // A fragments from LDS (contiguous k order, same as B: lanes<16 get K 0..15, lanes>=16 get K 16..31)
    v16h A[KCH];
    #pragma unroll
    for (int kc = 0; kc < KCH; ++kc) {
      const int kbase = kc * 32 + khalf * 16;
      const v8h lo = *(const v8h*)&a[col][kbase];
      const v8h hi = *(const v8h*)&a[col][kbase + 8];
      #pragma unroll
      for (int q = 0; q < 8; ++q) { A[kc][q] = lo[q]; A[kc][8 + q] = hi[q]; }
    }

    // z = [x_t | h | 1] @ [W | b]^T : one 16x16 WMMA tile per gate per K chunk.
    // First chunk uses constant-zero C (inline 0 operand), bias rides in B.
    v8f acc[4];
    #pragma unroll
    for (int g = 0; g < 4; ++g)
      acc[g] = __builtin_amdgcn_wmma_f32_16x16x32_f16(
          false, A[0], false, Bf[0][g], (short)0, zeroC, false, false);
    #pragma unroll
    for (int kc = 1; kc < KCH; ++kc) {
      #pragma unroll
      for (int g = 0; g < 4; ++g)
        acc[g] = __builtin_amdgcn_wmma_f32_16x16x32_f16(
            false, A[kc], false, Bf[kc][g], (short)0, acc[g], false, false);
    }

    // LSTM cell (gate order i, f, g, o) — all v_tanh based
    #pragma unroll
    for (int j = 0; j < 8; ++j) {
      const float ig = fast_sigmoid(acc[0][j]);
      const float fg = fast_sigmoid(acc[1][j]);
      const float gg = fast_tanh(acc[2][j]);
      const float og = fast_sigmoid(acc[3][j]);
      const float cc = __builtin_fmaf(fg, c_st[j], ig * gg);
      c_st[j] = cc;
      h_st[j] = og * fast_tanh(cc);
    }

    // store h_t (lanes with col<14; lane-contiguous within a row -> coalesced)
    if (col < HH) {
      #pragma unroll
      for (int j = 0; j < 8; ++j) {
        const size_t r = (size_t)(row0 + khalf * 8 + j);
        const size_t o = (r * TT + t) * (size_t)dstStride + dstOff + col;
        if (DST_F32) dstF[o] = h_st[j];
        else         dstH[o] = (_Float16)h_st[j];
      }
    }
  }
}

extern "C" void kernel_launch(void* const* d_in, const int* in_sizes, int n_in,
                              void* d_out, int out_size, void* d_ws, size_t ws_size,
                              hipStream_t stream)
{
  (void)in_sizes; (void)n_in; (void)out_size; (void)ws_size;

  const float* x     = (const float*)d_in[0];
  const float* wih1f = (const float*)d_in[1];
  const float* whh1f = (const float*)d_in[2];
  const float* bih1f = (const float*)d_in[3];
  const float* bhh1f = (const float*)d_in[4];
  const float* wih1b = (const float*)d_in[5];
  const float* whh1b = (const float*)d_in[6];
  const float* bih1b = (const float*)d_in[7];
  const float* bhh1b = (const float*)d_in[8];
  const float* wih2f = (const float*)d_in[9];
  const float* whh2f = (const float*)d_in[10];
  const float* bih2f = (const float*)d_in[11];
  const float* bhh2f = (const float*)d_in[12];
  const float* wih2b = (const float*)d_in[13];
  const float* whh2b = (const float*)d_in[14];
  const float* bih2b = (const float*)d_in[15];
  const float* bhh2b = (const float*)d_in[16];
  const float* wih3  = (const float*)d_in[17];
  const float* whh3  = (const float*)d_in[18];
  const float* bih3  = (const float*)d_in[19];
  const float* bhh3  = (const float*)d_in[20];

  // fp16 staging of the bidirectional layer outputs in scratch
  _Float16* y1 = (_Float16*)d_ws;                       // [B][T][28]
  _Float16* y2 = y1 + (size_t)BB * TT * 28;             // [B][T][28]

  const int nTiles = BB / 16;                           // 1024 wave-tiles
  dim3 blk(WPB * 32);
  dim3 grd((nTiles + WPB - 1) / WPB);

  // layer 1 (in=13, K=27 -> 1 chunk): forward then backward halves of y1
  lstm_pass<13,1,true ,false><<<grd,blk,0,stream>>>(x,  13, wih1f, whh1f, bih1f, bhh1f, y1, 28, 0,  0, nTiles);
  lstm_pass<13,1,true ,false><<<grd,blk,0,stream>>>(x,  13, wih1b, whh1b, bih1b, bhh1b, y1, 28, 14, 1, nTiles);
  // layer 2 (in=28, K=42 -> 2 chunks)
  lstm_pass<28,2,false,false><<<grd,blk,0,stream>>>(y1, 28, wih2f, whh2f, bih2f, bhh2f, y2, 28, 0,  0, nTiles);
  lstm_pass<28,2,false,false><<<grd,blk,0,stream>>>(y1, 28, wih2b, whh2b, bih2b, bhh2b, y2, 28, 14, 1, nTiles);
  // layer 3 (in=28), fp32 output straight to d_out
  lstm_pass<28,2,false,true ><<<grd,blk,0,stream>>>(y2, 28, wih3,  whh3,  bih3,  bhh3,  d_out, 14, 0, 0, nTiles);
}
